// GraphAttentionEncoder_80109730005641
// MI455X (gfx1250) — compile-verified, matrix operand order checked by reading, not run on hardware
//
#include <hip/hip_runtime.h>
#include <hip/hip_bf16.h>
#include <stdint.h>

#define NN 30000
#define KN 16
#define DD 128
#define ZZ 32
#define STR 136          // bf16 LDS row stride (272B: rows land on distinct banks)
#define FSTR 132         // f32 LDS row stride
#define LN_EPS 1e-5f
#define NEG_SLOPE 0.01f
#define INV_RAD (1.0f/(1.0f+1e-8f))

typedef __attribute__((ext_vector_type(16))) __bf16 v16bf;
typedef __attribute__((ext_vector_type(8)))  float  v8f;

__device__ __forceinline__ float wsum(float v) {
#pragma unroll
  for (int m = 16; m >= 1; m >>= 1) v += __shfl_xor(v, m, 32);
  return v;
}

// Build a 16x32-bf16 WMMA A/B fragment from two contiguous 16B chunks.
// Per-lane layout (ISA 7.12.2): lane<16 -> K = [0..7]+kb, [16..23]+kb; lane>=16 -> +8.
__device__ __forceinline__ v16bf ldfrag(const __bf16* p0, const __bf16* p1) {
  union { v16bf v; uint4 q[2]; } u;
  u.q[0] = *(const uint4*)p0;
  u.q[1] = *(const uint4*)p1;
  return u.v;
}

// ---------------------------------------------------------------------------
// Prep: fold LN affine into weights (bf16) and biases, convert W_red to bf16.
//   Wf[m] = W ∘ g (columns scaled), biasf[m] = b_lin + W @ b_ln
//   m: 0=Ws1(ln1) 1=Wn1(ln1) 2=Ws2(ln2) 3=Wn2(ln2)
// ---------------------------------------------------------------------------
__global__ void prep_weights(
    const float* __restrict__ Ws1, const float* __restrict__ Wn1,
    const float* __restrict__ Ws2, const float* __restrict__ Wn2,
    const float* __restrict__ g1,  const float* __restrict__ b1,
    const float* __restrict__ g2,  const float* __restrict__ b2,
    const float* __restrict__ bs1, const float* __restrict__ bn1,
    const float* __restrict__ bs2, const float* __restrict__ bn2,
    const float* __restrict__ Wred,
    __bf16* __restrict__ wf, __bf16* __restrict__ wredf,
    float* __restrict__ biasf)
{
  __shared__ float red[128];
  const int t = threadIdx.x;
  const int blk = blockIdx.x;
  if (blk < 512) {
    const int m = blk >> 7, e = blk & 127;
    const float* W; const float* blin;
    if      (m == 0) { W = Ws1; blin = bs1; }
    else if (m == 1) { W = Wn1; blin = bn1; }
    else if (m == 2) { W = Ws2; blin = bs2; }
    else             { W = Wn2; blin = bn2; }
    const float* g  = (m < 2) ? g1 : g2;
    const float* bl = (m < 2) ? b1 : b2;
    const float wv = W[e*128 + t];
    wf[m*16384 + e*128 + t] = (__bf16)(wv * g[t]);
    red[t] = wv * bl[t];
    __syncthreads();
    for (int s = 64; s >= 1; s >>= 1) {
      if (t < s) red[t] += red[t + s];
      __syncthreads();
    }
    if (t == 0) biasf[m*128 + e] = blin[e] + red[0];
  } else {
    const int e = blk - 512;                       // W_red rows (32)
    wredf[e*128 + t] = (__bf16)Wred[e*128 + t];
  }
}

// ---------------------------------------------------------------------------
// One attention layer for a 16-node tile (16 waves, wave w == node w).
// ---------------------------------------------------------------------------
__device__ __forceinline__ void attn_layer(
    const __bf16* __restrict__ Wsf, const float* __restrict__ bsf,
    const __bf16* __restrict__ Wnf, const float* __restrict__ bnf,
    const float* __restrict__ gRaw, const float* __restrict__ bRaw,
    float beta,
    __bf16* nbHat, __bf16* tHat, float* selfS, float* hOut,
    const float* distW, float* lnG, float* lnB)
{
  const int w    = threadIdx.x >> 5;
  const int lane = threadIdx.x & 31;
  const int c16  = lane & 15;
  const int kb0  = (lane & 16) ? 8 : 0;

  if (threadIdx.x < 128) { lnG[threadIdx.x] = gRaw[threadIdx.x]; lnB[threadIdx.x] = bRaw[threadIdx.x]; }

  // ---- self scores: tHat[16x128] @ Wsf^T, waves 0..7 take one 16-col chunk ----
  if (w < 8) {
    const int ecol = w*16 + c16;
    v8f acc = {};
#pragma unroll
    for (int ks = 0; ks < 4; ++ks) {
      const int kb = ks*32 + kb0;
      v16bf a  = ldfrag(&tHat[c16*STR + kb], &tHat[c16*STR + kb + 16]);
      v16bf bm = ldfrag(&Wsf[ecol*128 + kb], &Wsf[ecol*128 + kb + 16]);
      acc = __builtin_amdgcn_wmma_f32_16x16x32_bf16(false, a, false, bm, (short)0, acc, false, false);
    }
    const float bias = bsf[ecol];
#pragma unroll
    for (int v = 0; v < 8; ++v) {
      const int M = v + ((lane >> 4) << 3);
      selfS[M*FSTR + ecol] = acc[v] + bias;            // selfS[node][dim]
    }
  }
  __syncthreads();

  // ---- per-node: neighbor scores (WMMA, M = neighbor k) + softmax + context ----
  const int node = w;
#pragma unroll 1
  for (int chunk = 0; chunk < 8; ++chunk) {
    const int ecol = chunk*16 + c16;
    v8f acc = {};
#pragma unroll
    for (int ks = 0; ks < 4; ++ks) {
      const int kb = ks*32 + kb0;
      const __bf16* ar = &nbHat[(node*16 + c16)*STR + kb];
      v16bf a  = ldfrag(ar, ar + 16);
      v16bf bm = ldfrag(&Wnf[ecol*128 + kb], &Wnf[ecol*128 + kb + 16]);
      acc = __builtin_amdgcn_wmma_f32_16x16x32_bf16(false, a, false, bm, (short)0, acc, false, false);
    }
    const float biasn = bnf[ecol];
    const float gg = lnG[ecol], bb = lnB[ecol];
    const float s0 = selfS[node*FSTR + ecol];

    float sc[8];
    float mx = s0;
#pragma unroll
    for (int v = 0; v < 8; ++v) {
      const int M = v + ((lane >> 4) << 3);
      sc[v] = (acc[v] + biasn) * distW[node*16 + M];   // dist-weighted neighbor score
      mx = fmaxf(mx, sc[v]);
    }
    mx = fmaxf(mx, __shfl_xor(mx, 16, 32));            // column max over all 16 k + self

    float psum = 0.f, wctx = 0.f;
#pragma unroll
    for (int v = 0; v < 8; ++v) {
      const int M = v + ((lane >> 4) << 3);
      const float p = __expf(sc[v] - mx);
      psum += p;
      const float nbv = (float)nbHat[(node*16 + M)*STR + ecol];
      wctx += p * (nbv * gg + bb);                     // apply LN affine on the fly
    }
    const float psumT = psum + __shfl_xor(psum, 16, 32);
    const float wctxT = wctx + __shfl_xor(wctx, 16, 32);
    const float p0    = __expf(s0 - mx);
    const float denom = p0 + psumT;
    const float tAff  = (float)tHat[node*STR + ecol] * gg + bb;
    float ctx = beta * (p0/denom) * tAff + (1.f - beta) * (wctxT/denom);
    ctx = (ctx > 0.f) ? ctx : NEG_SLOPE*ctx;           // leaky relu
    if (lane < 16) hOut[node*FSTR + ecol] = ctx;
  }
  __syncthreads();
}

// ---------------------------------------------------------------------------
// Fused kernel: one 16-node tile per workgroup, 16 waves (512 threads).
// ---------------------------------------------------------------------------
__global__ __launch_bounds__(512, 1) void graph_attn_kernel(
    const float* __restrict__ x, const int* __restrict__ edge,
    const float* __restrict__ coords,
    const __bf16* __restrict__ wf, const float* __restrict__ biasf,
    const __bf16* __restrict__ wredf,
    const float* __restrict__ g1, const float* __restrict__ b1,
    const float* __restrict__ g2, const float* __restrict__ b2,
    const float* __restrict__ bred,
    float* __restrict__ out)
{
  __shared__ __bf16 nbHat[256*STR];   // normalized neighbors (shared by both layers)
  __shared__ __bf16 tHat[16*STR];     // normalized target (per layer)
  __shared__ float  selfS[16*FSTR];
  __shared__ float  hBuf[16*FSTR];
  __shared__ float  distW[16*16];
  __shared__ float  lnG[128], lnB[128];

  const int w    = threadIdx.x >> 5;
  const int lane = threadIdx.x & 31;
  const int tile = blockIdx.x;
  const int node = tile*16 + w;
  const int* colIdx = edge + (long long)NN*KN;          // second row of edge_index

  // ---- gather neighbors, per-row LN (normalize only), aggregate, distances ----
  float4 agg = make_float4(0.f, 0.f, 0.f, 0.f);
  const float2 pc = ((const float2*)coords)[node];
  for (int k = 0; k < KN; ++k) {
    const int c = colIdx[node*KN + k];
    const float4 v = ((const float4*)(x + (long long)c*DD))[lane];
    const float s  = wsum(v.x + v.y + v.z + v.w);
    const float sq = wsum(v.x*v.x + v.y*v.y + v.z*v.z + v.w*v.w);
    const float mean = s * (1.f/DD);
    const float rstd = rsqrtf(sq*(1.f/DD) - mean*mean + LN_EPS);
    __bf16* dst = &nbHat[(w*16 + k)*STR + lane*4];
    dst[0] = (__bf16)((v.x - mean)*rstd);
    dst[1] = (__bf16)((v.y - mean)*rstd);
    dst[2] = (__bf16)((v.z - mean)*rstd);
    dst[3] = (__bf16)((v.w - mean)*rstd);
    agg.x += v.x; agg.y += v.y; agg.z += v.z; agg.w += v.w;
    if (lane == 0) {
      const float2 qc = ((const float2*)coords)[c];
      const float dx = pc.x - qc.x, dy = pc.y - qc.y;
      distW[w*16 + k] = __expf(-sqrtf(dx*dx + dy*dy) * INV_RAD);
    }
  }
  { // LN(x_agg) -> tHat (layer 1 target)
    const float s  = wsum(agg.x + agg.y + agg.z + agg.w);
    const float sq = wsum(agg.x*agg.x + agg.y*agg.y + agg.z*agg.z + agg.w*agg.w);
    const float mean = s * (1.f/DD);
    const float rstd = rsqrtf(sq*(1.f/DD) - mean*mean + LN_EPS);
    __bf16* dst = &tHat[w*STR + lane*4];
    dst[0] = (__bf16)((agg.x - mean)*rstd);
    dst[1] = (__bf16)((agg.y - mean)*rstd);
    dst[2] = (__bf16)((agg.z - mean)*rstd);
    dst[3] = (__bf16)((agg.w - mean)*rstd);
  }
  __syncthreads();

  // ---- layer 1 ----
  attn_layer(wf, biasf, wf + 16384, biasf + 128, g1, b1, 0.5f,
             nbHat, tHat, selfS, hBuf, distW, lnG, lnB);

  // ---- LN(h1) -> tHat (layer 2 target) ----
  {
    const float v0 = hBuf[w*FSTR + lane*4 + 0];
    const float v1 = hBuf[w*FSTR + lane*4 + 1];
    const float v2 = hBuf[w*FSTR + lane*4 + 2];
    const float v3 = hBuf[w*FSTR + lane*4 + 3];
    const float s  = wsum(v0 + v1 + v2 + v3);
    const float sq = wsum(v0*v0 + v1*v1 + v2*v2 + v3*v3);
    const float mean = s * (1.f/DD);
    const float rstd = rsqrtf(sq*(1.f/DD) - mean*mean + LN_EPS);
    __bf16* dst = &tHat[w*STR + lane*4];
    dst[0] = (__bf16)((v0 - mean)*rstd);
    dst[1] = (__bf16)((v1 - mean)*rstd);
    dst[2] = (__bf16)((v2 - mean)*rstd);
    dst[3] = (__bf16)((v3 - mean)*rstd);
  }
  __syncthreads();

  // ---- layer 2 ----
  attn_layer(wf + 32768, biasf + 256, wf + 49152, biasf + 384, g2, b2, 0.5f,
             nbHat, tHat, selfS, hBuf, distW, lnG, lnB);

  // ---- h2 -> bf16 (reuse tHat) for final projection ----
  {
    __bf16* dst = &tHat[w*STR + lane*4];
    dst[0] = (__bf16)hBuf[w*FSTR + lane*4 + 0];
    dst[1] = (__bf16)hBuf[w*FSTR + lane*4 + 1];
    dst[2] = (__bf16)hBuf[w*FSTR + lane*4 + 2];
    dst[3] = (__bf16)hBuf[w*FSTR + lane*4 + 3];
  }
  __syncthreads();

  // ---- out = h2 @ W_red^T + b_red : waves 0..1, one 16-col chunk each ----
  if (w < 2) {
    const int c16 = lane & 15;
    const int kb0 = (lane & 16) ? 8 : 0;
    const int ecol = w*16 + c16;
    v8f acc = {};
#pragma unroll
    for (int ks = 0; ks < 4; ++ks) {
      const int kb = ks*32 + kb0;
      v16bf a  = ldfrag(&tHat[c16*STR + kb], &tHat[c16*STR + kb + 16]);
      v16bf bm = ldfrag(&wredf[ecol*128 + kb], &wredf[ecol*128 + kb + 16]);
      acc = __builtin_amdgcn_wmma_f32_16x16x32_bf16(false, a, false, bm, (short)0, acc, false, false);
    }
    const float bias = bred[ecol];
#pragma unroll
    for (int v = 0; v < 8; ++v) {
      const int M = v + ((lane >> 4) << 3);
      out[(long long)(tile*16 + M)*ZZ + ecol] = acc[v] + bias;
    }
  }
}

// ---------------------------------------------------------------------------
extern "C" void kernel_launch(void* const* d_in, const int* in_sizes, int n_in,
                              void* d_out, int out_size, void* d_ws, size_t ws_size,
                              hipStream_t stream) {
  const float* x      = (const float*)d_in[0];
  const int*   edge   = (const int*)  d_in[1];
  const float* coords = (const float*)d_in[2];
  const float* ln1_g  = (const float*)d_in[3];
  const float* ln1_b  = (const float*)d_in[4];
  const float* Ws1    = (const float*)d_in[5];
  const float* bs1    = (const float*)d_in[6];
  const float* Wn1    = (const float*)d_in[7];
  const float* bn1    = (const float*)d_in[8];
  const float* ln2_g  = (const float*)d_in[9];
  const float* ln2_b  = (const float*)d_in[10];
  const float* Ws2    = (const float*)d_in[11];
  const float* bs2    = (const float*)d_in[12];
  const float* Wn2    = (const float*)d_in[13];
  const float* bn2    = (const float*)d_in[14];
  const float* Wred   = (const float*)d_in[15];
  const float* bred   = (const float*)d_in[16];

  __bf16* wf    = (__bf16*)d_ws;                              // 4 * 128*128 bf16
  __bf16* wredf = (__bf16*)((char*)d_ws + 131072);            // 32*128 bf16
  float*  biasf = (float*) ((char*)d_ws + 139264);            // 4 * 128 f32

  prep_weights<<<544, 128, 0, stream>>>(Ws1, Wn1, Ws2, Wn2,
                                        ln1_g, ln1_b, ln2_g, ln2_b,
                                        bs1, bn1, bs2, bn2, Wred,
                                        wf, wredf, biasf);
  graph_attn_kernel<<<NN/16, 512, 0, stream>>>(x, edge, coords,
                                               wf, biasf, wredf,
                                               ln1_g, ln1_b, ln2_g, ln2_b,
                                               bred, (float*)d_out);
}
